// CNNxCNN_Attention_33990371181531
// MI455X (gfx1250) — matrix operand order repaired and checked by reading, not compiled, not run on hardware
//
#include <hip/hip_runtime.h>
#include <hip/hip_bf16.h>

// ---------------------------------------------------------------------------
// MI455X (gfx1250) implementation of block channel-attention.
// All GEMMs use v_wmma_f32_16x16x32_bf16 (wave32, 16x16 tile per wave).
// Intermediates stored bf16 (halve HBM traffic); accumulation in f32.
// L2-normalization folded into softmax via per-row sumsq (no extra pass).
// k3 stages q/k tiles with GLOBAL_LOAD_ASYNC_TO_LDS_B128 (ASYNCcnt path).
// ---------------------------------------------------------------------------

typedef __attribute__((ext_vector_type(8)))  float  v8f;
typedef __attribute__((ext_vector_type(16))) __bf16 v16bf;

#define BATCH 2
#define DIMC  48
#define QKVC  144
#define HEADS 8
#define CPH   6          // channels per head
#define CB    96         // c * 4 * 4  (blocked row dim)
#define IMG   512
#define HW    (512 * 512)
#define NB    16384      // 128*128 block-grid positions
#define H1    128

__device__ __forceinline__ v8f wmma_bf16(v16bf a, v16bf b, v8f c) {
  return __builtin_amdgcn_wmma_f32_16x16x32_bf16(false, a, false, b, (short)0, c, false, false);
}

// A-matrix fragment, 16x32 bf16 (ISA 7.12.2 layout): lane m = lane&15.
// VGPR v<4: K = 8*hi + 2v ; v>=4: K = 16 + 8*hi + 2(v-4).  Source: LDS, K-contiguous rows.
__device__ __forceinline__ v16bf fragA(const __bf16* base, int strideH, int kBase, int lane) {
  const __bf16* row = base + (lane & 15) * strideH;
  int hi8 = (lane >> 4) * 8;
  v16bf f;
#pragma unroll
  for (int v = 0; v < 8; ++v) {
    int k = kBase + ((v < 4) ? (hi8 + 2 * v) : (16 + hi8 + 2 * (v - 4)));
    f[2 * v]     = row[k];
    f[2 * v + 1] = row[k + 1];
  }
  return f;
}

// B-matrix fragment, 32x16 bf16: lane n = lane&15; VGPR v<4: K = 8*hi + 2v ; v>=4: K = 16 + 8*hi + 2(v-4).
// Source: LDS stored transposed ("row" indexed by n, K-contiguous).
__device__ __forceinline__ v16bf fragB(const __bf16* base, int strideH, int kBase, int lane) {
  const __bf16* row = base + (lane & 15) * strideH;
  int hi8 = (lane >> 4) * 8;
  v16bf f;
#pragma unroll
  for (int v = 0; v < 8; ++v) {
    int k = kBase + ((v < 4) ? (hi8 + 2 * v) : (16 + hi8 + 2 * (v - 4)));
    f[2 * v]     = row[k];
    f[2 * v + 1] = row[k + 1];
  }
  return f;
}

// ---------------------------------------------------------------------------
// Kernel 0: zero the atomic-accumulated buffers (sumsq_q, sumsq_k, gram)
// ---------------------------------------------------------------------------
__global__ void kzero(float* p, long n) {
  long i = (long)blockIdx.x * blockDim.x + threadIdx.x;
  if (i < n) p[i] = 0.f;
}

// ---------------------------------------------------------------------------
// Kernel 1: qkv 1x1 conv = GEMM  [144 x 48] * [48 x HW]  -> bf16 qkv
// grid (HW/128, 9, B); block 256 (8 waves); each wave one 16x16 tile.
// K=48 zero-padded to 64 (2 k-steps of 32).
// ---------------------------------------------------------------------------
#define KPS 66   // padded LDS row stride (odd dword count -> conflict-free)
__global__ __launch_bounds__(256) void k1_qkv(const float* __restrict__ x,
                                              const float* __restrict__ wq,
                                              __bf16* __restrict__ qkv) {
  __shared__ __bf16 A[16 * KPS];    // [m][k]  weights for this M tile
  __shared__ __bf16 xT[128 * KPS];  // [p_local][k] transposed activations
  int b = blockIdx.z, mt = blockIdx.y, p0 = blockIdx.x * 128;
  int t = threadIdx.x;

  for (int i = t; i < 16 * KPS; i += 256) {
    int m = i / KPS, k = i % KPS;
    A[i] = (k < DIMC) ? (__bf16)wq[(mt * 16 + m) * DIMC + k] : (__bf16)0.f;
  }
  for (int i = t; i < DIMC * 128; i += 256) {        // coalesced over pixels
    int ci = i >> 7, p = i & 127;
    const float* gp = &x[((long)b * DIMC + ci) * HW + p0 + p];
    xT[p * KPS + ci] = (__bf16)(*gp);
    __builtin_prefetch(gp + 128, 0, 3);              // global_prefetch_b8: warm next chunk
  }
  for (int i = t; i < 128 * (KPS - DIMC); i += 256) { // zero K pad
    int p = i / (KPS - DIMC), k = DIMC + i % (KPS - DIMC);
    xT[p * KPS + k] = (__bf16)0.f;
  }
  __syncthreads();

  int lane = t & 31, w = t >> 5;
  v8f acc = {0.f, 0.f, 0.f, 0.f, 0.f, 0.f, 0.f, 0.f};
#pragma unroll
  for (int ks = 0; ks < 2; ++ks) {
    v16bf a  = fragA(A, KPS, ks * 32, lane);
    v16bf bb = fragB(xT + w * 16 * KPS, KPS, ks * 32, lane);
    acc = wmma_bf16(a, bb, acc);
  }
  int n = p0 + w * 16 + (lane & 15);
  int mb = mt * 16 + ((lane >> 4) << 3);
#pragma unroll
  for (int r = 0; r < 8; ++r)
    qkv[((long)b * QKVC + mb + r) * HW + n] = (__bf16)acc[r];
}

// ---------------------------------------------------------------------------
// Kernel 2: 3x3 depthwise conv (zero pad) + scatter to blocked q/k/v (bf16)
//           + per-row sum-of-squares for q,k (LDS ds_add_f32, then atomics).
// grid (32, 32, B*144): 16x16 pixel tile per block; 18x18 halo staged in LDS.
// ---------------------------------------------------------------------------
__global__ __launch_bounds__(256) void k2_dw(const __bf16* __restrict__ qkv,
                                             const float* __restrict__ wd,
                                             __bf16* __restrict__ qb,
                                             __bf16* __restrict__ kb,
                                             __bf16* __restrict__ vb,
                                             float* __restrict__ ssq,
                                             float* __restrict__ ssk) {
  __shared__ __bf16 tile[18 * 20];
  __shared__ float red[16];
  int bc = blockIdx.z;
  int b = bc / QKVC, ch = bc % QKVC;
  int t = threadIdx.x;
  int tx = t & 15, ty = t >> 4;
  int gx0 = blockIdx.x * 16, gy0 = blockIdx.y * 16;
  const __bf16* src = qkv + ((long)b * QKVC + ch) * HW;

  for (int i = t; i < 18 * 18; i += 256) {           // halo tile -> LDS once
    int ly = i / 18, lx = i % 18;
    int gy = gy0 + ly - 1, gx = gx0 + lx - 1;
    __bf16 val = (__bf16)0.f;
    if (gy >= 0 && gy < IMG && gx >= 0 && gx < IMG) val = src[gy * IMG + gx];
    tile[ly * 20 + lx] = val;
  }
  if (t < 16) red[t] = 0.f;
  __syncthreads();

  float wv[9];
#pragma unroll
  for (int i = 0; i < 9; ++i) wv[i] = wd[ch * 9 + i];

  float acc = 0.f;
#pragma unroll
  for (int dy = 0; dy < 3; ++dy)
#pragma unroll
    for (int dx = 0; dx < 3; ++dx)
      acc += wv[dy * 3 + dx] * (float)tile[(ty + dy) * 20 + (tx + dx)];

  int xx = gx0 + tx, yy = gy0 + ty;
  int g = ch % DIMC, grp = ch / DIMC;          // 0=q 1=k 2=v
  int head = g / CPH, c = g % CPH;
  int sub = (yy & 3) * 4 + (xx & 3);
  int row = c * 16 + sub;
  int col = (yy >> 2) * H1 + (xx >> 2);
  __bf16* dst = (grp == 0) ? qb : (grp == 1) ? kb : vb;
  dst[(((long)b * HEADS + head) * CB + row) * NB + col] = (__bf16)acc;

  if (grp < 2) atomicAdd(&red[sub], acc * acc);          // ds_add_f32
  __syncthreads();
  if (grp < 2 && t < 16) {
    float* s = (grp == 0) ? ssq : ssk;
    atomicAdd(&s[((b * HEADS + head) * CB) + c * 16 + t], red[t]);
  }
}

// ---------------------------------------------------------------------------
// Kernel 3: gram = q . k^T per (b,head): 96x96, K split into 128 chunks of 128,
// f32 atomic reduction into gram. grid (128, 16); block 256 (8 waves, 36 tiles).
// Staging via GLOBAL_LOAD_ASYNC_TO_LDS_B128 (ASYNCcnt), 16B-aligned LDS rows.
// ---------------------------------------------------------------------------
#define KC  128
#define KRS 136   // LDS row stride in halfs: 272 B, 16B-aligned rows
__global__ __launch_bounds__(256) void k3_gram(const __bf16* __restrict__ qb,
                                               const __bf16* __restrict__ kb,
                                               float* __restrict__ G) {
  __shared__ __align__(16) __bf16 sQ[CB * KRS];
  __shared__ __align__(16) __bf16 sK[CB * KRS];
  int bh = blockIdx.y;
  int k0 = blockIdx.x * KC;
  int t = threadIdx.x;
  const __bf16* qbase = qb + (long)bh * CB * NB + k0;
  const __bf16* kbase = kb + (long)bh * CB * NB + k0;

  // 96 rows x 256 B per array = 1536 chunks of 16 B; direct memory->LDS DMA.
  for (int i = t; i < CB * (KC / 8); i += 256) {
    int r = i >> 4, d = i & 15;                       // row, 16B-chunk in row
    const __bf16* gq = qbase + (long)r * NB + d * 8;
    const __bf16* gk = kbase + (long)r * NB + d * 8;
    unsigned lq = (unsigned)(uintptr_t)(sQ + r * KRS + d * 8);
    unsigned lk = (unsigned)(uintptr_t)(sK + r * KRS + d * 8);
    asm volatile("global_load_async_to_lds_b128 %0, %1, off"
                 :: "v"(lq), "v"(gq) : "memory");
    asm volatile("global_load_async_to_lds_b128 %0, %1, off"
                 :: "v"(lk), "v"(gk) : "memory");
  }
  asm volatile("s_wait_asynccnt 0x0" ::: "memory");
  __syncthreads();

  int lane = t & 31, w = t >> 5;
  float* g = G + (long)bh * CB * CB;
  for (int tid = w; tid < 36; tid += 8) {
    int mt = tid / 6, nt = tid % 6;
    v8f acc = {0.f, 0.f, 0.f, 0.f, 0.f, 0.f, 0.f, 0.f};
#pragma unroll
    for (int ks = 0; ks < KC / 32; ++ks) {
      v16bf a  = fragA(sQ + mt * 16 * KRS, KRS, ks * 32, lane);
      v16bf bb = fragB(sK + nt * 16 * KRS, KRS, ks * 32, lane);
      acc = wmma_bf16(a, bb, acc);
    }
    int n = nt * 16 + (lane & 15);
    int mb = mt * 16 + ((lane >> 4) << 3);
#pragma unroll
    for (int r = 0; r < 8; ++r)
      atomicAdd(&g[(mb + r) * CB + n], acc[r]);        // global_atomic_add_f32
  }
}

// ---------------------------------------------------------------------------
// Kernel 4: fold normalization + temperature, softmax rows -> bf16 attn.
// grid(16); block 128. Tiny.
// ---------------------------------------------------------------------------
__global__ __launch_bounds__(128) void k4_softmax(const float* __restrict__ G,
                                                  const float* __restrict__ ssq,
                                                  const float* __restrict__ ssk,
                                                  const float* __restrict__ temp,
                                                  __bf16* __restrict__ attn) {
  __shared__ float nq[CB], nk[CB], red[128];
  int bh = blockIdx.x;
  float T = temp[bh & 7];
  int t = threadIdx.x;
  if (t < CB) {
    nq[t] = fmaxf(sqrtf(ssq[bh * CB + t]), 1e-12f);
    nk[t] = fmaxf(sqrtf(ssk[bh * CB + t]), 1e-12f);
  }
  __syncthreads();
  for (int i = 0; i < CB; ++i) {
    float v = -3.0e38f;
    if (t < CB) v = G[((long)bh * CB + i) * CB + t] * T / (nq[i] * nk[t]);
    red[t] = v; __syncthreads();
    for (int s = 64; s > 0; s >>= 1) { if (t < s) red[t] = fmaxf(red[t], red[t + s]); __syncthreads(); }
    float mx = red[0]; __syncthreads();
    float e = (t < CB) ? __expf(v - mx) : 0.f;
    red[t] = e; __syncthreads();
    for (int s = 64; s > 0; s >>= 1) { if (t < s) red[t] += red[t + s]; __syncthreads(); }
    float sum = red[0]; __syncthreads();
    if (t < CB) attn[((long)bh * CB + i) * CB + t] = (__bf16)(e / sum);
  }
}

// ---------------------------------------------------------------------------
// Kernel 5: out_blk = attn[96x96] @ v[96x16384], fused un-blocking store (bf16).
// grid (16384/128, 16); block 256. K=96 exactly 3 k-steps.
// ---------------------------------------------------------------------------
#define ARS 98
__global__ __launch_bounds__(256) void k5_av(const __bf16* __restrict__ attn,
                                             const __bf16* __restrict__ vb,
                                             __bf16* __restrict__ aout) {
  __shared__ __bf16 As[CB * ARS];    // attn [m][k]
  __shared__ __bf16 vT[128 * ARS];   // [n_local][k]  (transposed v chunk)
  int bh = blockIdx.y;
  int b = bh >> 3, head = bh & 7;
  int n0 = blockIdx.x * 128;
  int t = threadIdx.x;

  for (int i = t; i < CB * CB; i += 256) {
    int m = i / CB, k = i % CB;
    As[m * ARS + k] = attn[((long)bh * CB + m) * CB + k];
  }
  for (int i = t; i < CB * 128; i += 256) {           // coalesced over n
    int k = i >> 7, n = i & 127;
    vT[n * ARS + k] = vb[((long)bh * CB + k) * NB + n0 + n];
  }
  __syncthreads();

  int lane = t & 31, w = t >> 5;
  int nloc = w * 16 + (lane & 15);
  int col = n0 + nloc;
  int a1 = col >> 7, bw = col & 127;
  int hi8 = (lane >> 4) * 8;
  for (int mt = 0; mt < 6; ++mt) {
    v8f acc = {0.f, 0.f, 0.f, 0.f, 0.f, 0.f, 0.f, 0.f};
#pragma unroll
    for (int ks = 0; ks < 3; ++ks) {
      v16bf a  = fragA(As + mt * 16 * ARS, ARS, ks * 32, lane);
      v16bf bb = fragB(vT + w * 16 * ARS, ARS, ks * 32, lane);
      acc = wmma_bf16(a, bb, acc);
    }
#pragma unroll
    for (int r = 0; r < 8; ++r) {
      int row = mt * 16 + hi8 + r;
      int c = row >> 4, sub = row & 15;
      int hh = a1 * 4 + (sub >> 2);
      int ww = bw * 4 + (sub & 3);
      int chn = head * CPH + c;
      aout[(((long)b * DIMC + chn) * IMG + hh) * IMG + ww] = (__bf16)acc[r];
    }
  }
}

// ---------------------------------------------------------------------------
// Kernel 6: proj 1x1 conv = GEMM [48 x 48] * [48 x HW] -> f32 output.
// grid (HW/128, 3, B); block 256. K=48 padded to 64.
// ---------------------------------------------------------------------------
__global__ __launch_bounds__(256) void k6_proj(const __bf16* __restrict__ ain,
                                               const float* __restrict__ wp,
                                               float* __restrict__ out) {
  __shared__ __bf16 A[16 * KPS];
  __shared__ __bf16 xT[128 * KPS];
  int b = blockIdx.z, mt = blockIdx.y, p0 = blockIdx.x * 128;
  int t = threadIdx.x;

  for (int i = t; i < 16 * KPS; i += 256) {
    int m = i / KPS, k = i % KPS;
    A[i] = (k < DIMC) ? (__bf16)wp[(mt * 16 + m) * DIMC + k] : (__bf16)0.f;
  }
  for (int i = t; i < DIMC * 128; i += 256) {
    int ci = i >> 7, p = i & 127;
    const __bf16* gp = &ain[((long)b * DIMC + ci) * HW + p0 + p];
    xT[p * KPS + ci] = *gp;
    __builtin_prefetch(gp + 128, 0, 3);              // warm next chunk in GL2
  }
  for (int i = t; i < 128 * (KPS - DIMC); i += 256) {
    int p = i / (KPS - DIMC), k = DIMC + i % (KPS - DIMC);
    xT[p * KPS + k] = (__bf16)0.f;
  }
  __syncthreads();

  int lane = t & 31, w = t >> 5;
  v8f acc = {0.f, 0.f, 0.f, 0.f, 0.f, 0.f, 0.f, 0.f};
#pragma unroll
  for (int ks = 0; ks < 2; ++ks) {
    v16bf a  = fragA(A, KPS, ks * 32, lane);
    v16bf bb = fragB(xT + w * 16 * KPS, KPS, ks * 32, lane);
    acc = wmma_bf16(a, bb, acc);
  }
  int n = p0 + w * 16 + (lane & 15);
  int mb = mt * 16 + ((lane >> 4) << 3);
#pragma unroll
  for (int r = 0; r < 8; ++r)
    out[((long)b * DIMC + mb + r) * HW + n] = acc[r];
}

// ---------------------------------------------------------------------------
extern "C" void kernel_launch(void* const* d_in, const int* in_sizes, int n_in,
                              void* d_out, int out_size, void* d_ws, size_t ws_size,
                              hipStream_t stream) {
  const float* x     = (const float*)d_in[0];
  const float* qkv_w = (const float*)d_in[1];
  const float* dw_w  = (const float*)d_in[2];
  const float* proj_w= (const float*)d_in[3];
  const float* temp  = (const float*)d_in[4];
  float* out = (float*)d_out;
  char* ws = (char*)d_ws;

  size_t off = 0;
  auto take = [&](size_t bytes) { size_t cur = off; off += (bytes + 255) & ~(size_t)255; return cur; };

  size_t o_qkv  = take((size_t)BATCH * QKVC * HW * sizeof(__bf16));          // 151 MB
  size_t o_zero = off;
  size_t o_ssq  = take((size_t)BATCH * HEADS * CB * sizeof(float));
  size_t o_ssk  = take((size_t)BATCH * HEADS * CB * sizeof(float));
  size_t o_G    = take((size_t)BATCH * HEADS * CB * CB * sizeof(float));
  size_t zeroN  = (off - o_zero) / sizeof(float);
  size_t o_qb   = take((size_t)BATCH * HEADS * CB * NB * sizeof(__bf16));    // 50 MB
  size_t o_kb   = take((size_t)BATCH * HEADS * CB * NB * sizeof(__bf16));
  size_t o_vb   = take((size_t)BATCH * HEADS * CB * NB * sizeof(__bf16));
  size_t o_at   = take((size_t)BATCH * HEADS * CB * CB * sizeof(__bf16));
  size_t o_ao   = take((size_t)BATCH * DIMC * HW * sizeof(__bf16));          // 50 MB

  __bf16* qkv  = (__bf16*)(ws + o_qkv);
  float*  ssq  = (float*)(ws + o_ssq);
  float*  ssk  = (float*)(ws + o_ssk);
  float*  G    = (float*)(ws + o_G);
  __bf16* qb   = (__bf16*)(ws + o_qb);
  __bf16* kb   = (__bf16*)(ws + o_kb);
  __bf16* vb   = (__bf16*)(ws + o_vb);
  __bf16* attn = (__bf16*)(ws + o_at);
  __bf16* aout = (__bf16*)(ws + o_ao);

  kzero<<<dim3((unsigned)((zeroN + 255) / 256)), 256, 0, stream>>>((float*)(ws + o_zero), (long)zeroN);
  k1_qkv<<<dim3(HW / 128, QKVC / 16, BATCH), 256, 0, stream>>>(x, qkv_w, qkv);
  k2_dw<<<dim3(IMG / 16, IMG / 16, BATCH * QKVC), 256, 0, stream>>>(qkv, dw_w, qb, kb, vb, ssq, ssk);
  k3_gram<<<dim3(NB / KC, BATCH * HEADS), 256, 0, stream>>>(qb, kb, G);
  k4_softmax<<<dim3(BATCH * HEADS), 128, 0, stream>>>(G, ssq, ssk, temp, attn);
  k5_av<<<dim3(NB / 128, BATCH * HEADS), 256, 0, stream>>>(attn, vb, aout);
  k6_proj<<<dim3(HW / 128, DIMC / 16, BATCH), 256, 0, stream>>>(aout, proj_w, out);
}
